// Fingerprint_61211873902598
// MI455X (gfx1250) — compile-verified, matrix operand order checked by reading, not run on hardware
//
#include <hip/hip_runtime.h>
#include <cmath>

// ======================= CDNA5 / gfx1250 implementation ======================
// AttentiveFP-style GNN. Dense GEMMs use v_wmma_f32_16x16x32_f16 (f16 inputs,
// f32 accumulate), 2 N-tiles per wave (A-fragment reuse), straight-line
// global_load_b128 fragment loads with loop-invariant row pointers hoisted
// out of the K loop. Param leaves assumed in jax tree_flatten order.

typedef __attribute__((ext_vector_type(16))) _Float16 v16h;
typedef __attribute__((ext_vector_type(8)))  float    v8f;

static inline int cdiv(long a, long b) { return (int)((a + b - 1) / b); }

// ----------------------------- device helpers ------------------------------
__device__ __forceinline__ float act_eval(float y, int act) {
  switch (act) {
    case 1: return y > 0.f ? y : 0.f;             // relu
    case 2: return y > 0.f ? y : 0.01f * y;       // leaky_relu(0.01)
    case 3: return y > 0.f ? y : (expf(y) - 1.f); // elu
  }
  return y;
}

// monotonic float<->uint encoding so unsigned atomicMax == float max
__device__ __forceinline__ unsigned fenc(float f) {
  unsigned u = __float_as_uint(f);
  return (u & 0x80000000u) ? ~u : (u | 0x80000000u);
}
__device__ __forceinline__ float fdec(unsigned u) {
  return (u & 0x80000000u) ? __uint_as_float(u & 0x7FFFFFFFu) : __uint_as_float(~u);
}

__device__ __forceinline__ void put4(v16h& o, int base, float4 v) {
  o[base + 0] = (_Float16)v.x;
  o[base + 1] = (_Float16)v.y;
  o[base + 2] = (_Float16)v.z;
  o[base + 3] = (_Float16)v.w;
}

// Convert 16 contiguous-pair f32s at p (two 8-float runs, second at +16) into
// one 16x32-f16 WMMA fragment slice for this lane.
__device__ __forceinline__ v16h cvt_frag(const float* __restrict__ p) {
  float4 a0 = ((const float4*)p)[0];
  float4 a1 = ((const float4*)p)[1];
  float4 b0 = ((const float4*)(p + 16))[0];
  float4 b1 = ((const float4*)(p + 16))[1];
  v16h o;
  put4(o, 0, a0); put4(o, 4, a1); put4(o, 8, b0); put4(o, 12, b1);
  return o;
}

// --------------------------------- kernels ---------------------------------
// Y[N x Dout] = gather(X, idx)[N x Din] @ W^T + bias ; optional BN-stat
// atomics. Din % 32 == 0, Dout % 32 == 0. Each wave: 16 rows x 32 cols.
__global__ void gemm_wmma_kernel(const float* __restrict__ X, const int* __restrict__ idx,
                                 const float* __restrict__ W, const float* __restrict__ bias,
                                 float* __restrict__ Y,
                                 float* __restrict__ colsum, float* __restrict__ colsumsq,
                                 int N, int Din, int Dout) {
  int lane   = threadIdx.x & 31;
  int wave   = threadIdx.x >> 5;
  int tileM  = blockIdx.x * 8 + wave;
  int tileN2 = blockIdx.y;                 // pair of 16-wide column tiles
  if (tileM * 16 >= N) return;

  // Loop-invariant fragment row pointers (idx resolved once, rows clamped;
  // clamped rows' products are discarded in the epilogue).
  int koff = (lane >> 4) * 8;
  int rA = tileM * 16 + (lane & 15);
  rA = (rA < N) ? rA : (N - 1);
  long rrA = idx ? (long)idx[rA] : (long)rA;
  const float* pa  = X + rrA * (long)Din + koff;
  const float* pb0 = W + (long)(tileN2 * 32 + (lane & 15)) * Din + koff;
  const float* pb1 = pb0 + (long)16 * Din;

  v8f acc0 = {}, acc1 = {};
  for (int k0 = 0; k0 < Din; k0 += 32) {
    v16h fa  = cvt_frag(pa  + k0);
    v16h fb0 = cvt_frag(pb0 + k0);
    v16h fb1 = cvt_frag(pb1 + k0);
    acc0 = __builtin_amdgcn_wmma_f32_16x16x32_f16(false, fa, false, fb0,
                                                  (short)0, acc0, false, false);
    acc1 = __builtin_amdgcn_wmma_f32_16x16x32_f16(false, fa, false, fb1,
                                                  (short)0, acc1, false, false);
  }

  int hi = lane >> 4;
  int n0 = tileN2 * 32 + (lane & 15);
  int n1 = n0 + 16;
  float bv0 = bias ? bias[n0] : 0.f;
  float bv1 = bias ? bias[n1] : 0.f;
  float s0 = 0.f, q0 = 0.f, s1 = 0.f, q1 = 0.f;
#pragma unroll
  for (int r = 0; r < 8; ++r) {
    int row = tileM * 16 + hi * 8 + r;     // D layout: VGPR r -> M = r + 8*half
    if (row < N) {
      float y0 = acc0[r] + bv0;
      float y1 = acc1[r] + bv1;
      float* yr = Y + (size_t)row * Dout;
      yr[n0] = y0;
      yr[n1] = y1;
      s0 += y0; q0 += y0 * y0;
      s1 += y1; q1 += y1 * y1;
    }
  }
  if (colsum) {
    atomicAdd(&colsum[n0], s0);
    atomicAdd(&colsumsq[n0], q0);
    atomicAdd(&colsum[n1], s1);
    atomicAdd(&colsumsq[n1], q1);
  }
}

__global__ void fill_kernel(float* p, float v, size_t n) {
  size_t i = (size_t)blockIdx.x * blockDim.x + threadIdx.x;
  if (i < n) p[i] = v;
}

// dst[N x DinPad] = src[N x Din] zero-padded along K
__global__ void pad_kernel(const float* __restrict__ src, float* __restrict__ dst,
                           int N, int Din, int DinPad) {
  size_t i = (size_t)blockIdx.x * blockDim.x + threadIdx.x;
  if (i >= (size_t)N * DinPad) return;
  int k = (int)(i % (size_t)DinPad);
  size_t n = i / (size_t)DinPad;
  dst[i] = (k < Din) ? src[n * (size_t)Din + k] : 0.f;
}

// stat layout: [0..D) = sum, [D..2D) = sumsq, [2D..3D) = scale, [3D..4D) = shift
__global__ void bn_finalize_kernel(float* stat, const float* g, const float* be,
                                   float ninv, int Dout) {
  int j = blockIdx.x * blockDim.x + threadIdx.x;
  if (j >= Dout) return;
  float mean = stat[j] * ninv;
  float var  = stat[Dout + j] * ninv - mean * mean;
  float sc   = rsqrtf(var + 1e-6f) * g[j];
  stat[2 * Dout + j] = sc;
  stat[3 * Dout + j] = be[j] - mean * sc;
}

__global__ void bn_apply_kernel(float* Y, const float* stat, int Dout,
                                size_t total, int act) {
  size_t i = (size_t)blockIdx.x * blockDim.x + threadIdx.x;
  if (i >= total) return;
  int col = (int)(i % (size_t)Dout);
  float y = Y[i] * stat[2 * Dout + col] + stat[3 * Dout + col];
  Y[i] = act_eval(y, act);
}

__global__ void act_kernel(float* Y, size_t total, int act) {
  size_t i = (size_t)blockIdx.x * blockDim.x + threadIdx.x;
  if (i < total) Y[i] = act_eval(Y[i], act);
}

// Dout==1 GEMV over fa = [gather(P1,i1) | gather(P2,i2)] ; one wave per row.
// d1/d2 multiples of 4, rows 16B-aligned.
__global__ void dot2_kernel(const float* __restrict__ P1, const int* __restrict__ i1, int d1,
                            const float* __restrict__ P2, const int* __restrict__ i2, int d2,
                            const float* __restrict__ W, const float* __restrict__ b,
                            float* __restrict__ out, float* __restrict__ stat2, int M) {
  int wv   = (int)((blockIdx.x * (size_t)blockDim.x + threadIdx.x) >> 5);
  int lane = threadIdx.x & 31;
  if (wv >= M) return;
  const float* r1 = P1 + (size_t)(i1 ? i1[wv] : wv) * d1;
  float acc = 0.f;
  for (int k = lane * 4; k < d1; k += 128) {
    float4 x = *(const float4*)(r1 + k);
    float4 w = *(const float4*)(W + k);
    acc += x.x * w.x + x.y * w.y + x.z * w.z + x.w * w.w;
  }
  if (d2 > 0) {
    const float* r2 = P2 + (size_t)(i2 ? i2[wv] : wv) * d2;
    for (int k = lane * 4; k < d2; k += 128) {
      float4 x = *(const float4*)(r2 + k);
      float4 w = *(const float4*)(W + d1 + k);
      acc += x.x * w.x + x.y * w.y + x.z * w.z + x.w * w.w;
    }
  }
#pragma unroll
  for (int off = 16; off; off >>= 1) acc += __shfl_xor(acc, off, 32);
  if (lane == 0) {
    float y = acc + b[0];
    out[wv] = y;
    if (stat2) { atomicAdd(&stat2[0], y); atomicAdd(&stat2[1], y * y); }
  }
}

__global__ void seg_max_kernel(const float* x, const int* seg, unsigned* smax, int M) {
  int i = blockIdx.x * blockDim.x + threadIdx.x;
  if (i < M) atomicMax(&smax[seg[i]], fenc(x[i]));
}
__global__ void seg_expsum_kernel(const float* x, const int* seg, const unsigned* smax,
                                  float* w, float* ssum, int M) {
  int i = blockIdx.x * blockDim.x + threadIdx.x;
  if (i >= M) return;
  float e = expf(x[i] - fdec(smax[seg[i]]));
  w[i] = e;
  atomicAdd(&ssum[seg[i]], e);
}
__global__ void seg_norm_kernel(float* w, const int* seg, const float* ssum, int M) {
  int i = blockIdx.x * blockDim.x + threadIdx.x;
  if (i < M) w[i] = w[i] / (ssum[seg[i]] + 1e-8f);
}

// dst[seg[m], j] += (w ? w[m] : 1) * src[m, j]
__global__ void seg_scatter_kernel(const float* __restrict__ src, const float* __restrict__ w,
                                   const int* __restrict__ seg, float* __restrict__ dst,
                                   size_t total, int D) {
  size_t i = (size_t)blockIdx.x * blockDim.x + threadIdx.x;
  if (i >= total) return;
  size_t m = i / (size_t)D;
  int j = (int)(i % (size_t)D);
  float v = src[i] * (w ? w[m] : 1.0f);
  atomicAdd(&dst[(size_t)seg[m] * D + j], v);
}

// neighbor = [b | na | na + b - na*b] with na = a[bi1]
__global__ void neigh_kernel(const float* __restrict__ a, const float* __restrict__ b,
                             const int* __restrict__ bi1, float* __restrict__ neigh,
                             size_t total, int F) {
  size_t i = (size_t)blockIdx.x * blockDim.x + threadIdx.x;
  if (i >= total) return;
  size_t m = i / (size_t)F;
  int j = (int)(i % (size_t)F);
  float na = a[(size_t)bi1[m] * F + j];
  float bb = b[i];
  size_t base = m * (size_t)(3 * F) + j;
  neigh[base]         = bb;
  neigh[base + F]     = na;
  neigh[base + 2 * F] = na + bb - na * bb;
}

__global__ void split_idx_kernel(const int* bidx, int* b0, int* b1, int M) {
  int i = blockIdx.x * blockDim.x + threadIdx.x;
  if (i >= M) return;
  b0[i] = bidx[2 * i];
  b1[i] = bidx[2 * i + 1];
}

__global__ void gru_combine_kernel(const float* __restrict__ gi, const float* __restrict__ gh,
                                   const float* __restrict__ h, float* __restrict__ out,
                                   size_t total, int H) {
  size_t i = (size_t)blockIdx.x * blockDim.x + threadIdx.x;
  if (i >= total) return;
  size_t row = i / (size_t)H;
  int j = (int)(i % (size_t)H);
  const float* gir = gi + row * (size_t)(3 * H);
  const float* ghr = gh + row * (size_t)(3 * H);
  float r = 1.f / (1.f + expf(-(gir[j] + ghr[j])));
  float z = 1.f / (1.f + expf(-(gir[H + j] + ghr[H + j])));
  float n = tanhf(gir[2 * H + j] + r * ghr[2 * H + j]);
  out[i] = (1.f - z) * n + z * h[i];
}

// ------------------------------ host helpers -------------------------------
struct LinBNP { const float *W, *b, *be, *g; };

static void run_gemm(hipStream_t s, const float* X, const int* idx, int N, int Din, int Dout,
                     const float* W, const float* b, float* Y,
                     float* colsum, float* colsumsq) {
  dim3 grid(cdiv(cdiv(N, 16), 8), Dout / 32);
  gemm_wmma_kernel<<<grid, 256, 0, s>>>(X, idx, W, b, Y, colsum, colsumsq, N, Din, Dout);
}

static void run_linbn(hipStream_t s, const float* X, const int* idx, int N, int Din, int Dout,
                      const float* W, LinBNP p, float* Y, float* stat, int act) {
  fill_kernel<<<cdiv(2 * Dout, 256), 256, 0, s>>>(stat, 0.f, (size_t)(2 * Dout));
  run_gemm(s, X, idx, N, Din, Dout, W, p.b, Y, stat, stat + Dout);
  bn_finalize_kernel<<<cdiv(Dout, 64), 64, 0, s>>>(stat, p.g, p.be, 1.0f / (float)N, Dout);
  size_t total = (size_t)N * Dout;
  bn_apply_kernel<<<cdiv(total, 256), 256, 0, s>>>(Y, stat, Dout, total, act);
}

static void run_scores(hipStream_t s, const float* P1, const int* i1, int d1,
                       const float* P2, const int* i2, int d2,
                       LinBNP p, int M, const int* seg, int numseg,
                       float* score, float* wgt, float* stat,
                       unsigned* smax, float* ssum) {
  fill_kernel<<<1, 64, 0, s>>>(stat, 0.f, 4);
  dot2_kernel<<<cdiv(M, 8), 256, 0, s>>>(P1, i1, d1, P2, i2, d2, p.W, p.b, score, stat, M);
  bn_finalize_kernel<<<1, 64, 0, s>>>(stat, p.g, p.be, 1.0f / (float)M, 1);
  bn_apply_kernel<<<cdiv(M, 256), 256, 0, s>>>(score, stat, 1, (size_t)M, 2 /*leaky*/);
  fill_kernel<<<cdiv(numseg, 256), 256, 0, s>>>((float*)smax, 0.f, (size_t)numseg);
  fill_kernel<<<cdiv(numseg, 256), 256, 0, s>>>(ssum, 0.f, (size_t)numseg);
  seg_max_kernel<<<cdiv(M, 256), 256, 0, s>>>(score, seg, smax, M);
  seg_expsum_kernel<<<cdiv(M, 256), 256, 0, s>>>(score, seg, smax, wgt, ssum, M);
  seg_norm_kernel<<<cdiv(M, 256), 256, 0, s>>>(wgt, seg, ssum, M);
}

static void run_ctx(hipStream_t s, const float* att, const float* wgt, const int* seg,
                    float* ctx, int M, int Nseg, int D, bool elu) {
  size_t nz = (size_t)Nseg * D;
  fill_kernel<<<cdiv(nz, 256), 256, 0, s>>>(ctx, 0.f, nz);
  size_t total = (size_t)M * D;
  seg_scatter_kernel<<<cdiv(total, 256), 256, 0, s>>>(att, wgt, seg, ctx, total, D);
  if (elu) act_kernel<<<cdiv(nz, 256), 256, 0, s>>>(ctx, nz, 3);
}

static void run_gru(hipStream_t s, const float* x, const float* h,
                    const float* Wih, const float* bih,
                    const float* Whh, const float* bhh,
                    int N, int Din, int H, float* gi, float* gh, float* out) {
  run_gemm(s, x, nullptr, N, Din, 3 * H, Wih, bih, gi, nullptr, nullptr);
  run_gemm(s, h, nullptr, N, H,   3 * H, Whh, bhh, gh, nullptr, nullptr);
  size_t total = (size_t)N * H;
  gru_combine_kernel<<<cdiv(total, 256), 256, 0, s>>>(gi, gh, h, out, total, H);
}

// --------------------------------- launch ----------------------------------
extern "C" void kernel_launch(void* const* d_in, const int* in_sizes, int n_in,
                              void* d_out, int out_size, void* d_ws, size_t ws_size,
                              hipStream_t stream) {
  const int F = 128;
  const int NA = in_sizes[0] / 39;   // 100000 atoms
  const int NB = in_sizes[1] / 10;   // 400000 bonds
  const int NM = out_size;           // 5000 molecules

  const float* atom = (const float*)d_in[0];
  const float* bond = (const float*)d_in[1];
  const int* bidx   = (const int*)d_in[2];
  const int* molidx = (const int*)d_in[3];
  auto F32 = [&](int i) { return (const float*)d_in[i]; };

  // jax tree_flatten alphabetical leaf order
  LinBNP align      { F32(4),  F32(5),  F32(6),  F32(7)  };
  LinBNP atom_l1    { F32(8),  F32(9),  F32(10), F32(11) };
  LinBNP atom_l2    { F32(12), F32(13), F32(14), F32(15) };
  LinBNP attend     { F32(16), F32(17), F32(18), F32(19) };
  LinBNP bond_l1    { F32(20), F32(21), F32(22), F32(23) };
  LinBNP bond_l2    { F32(24), F32(25), F32(26), F32(27) };
  const float *gWhh = F32(28), *gWih = F32(29), *gbhh = F32(30), *gbih = F32(31);
  const float *predW = F32(32), *predB = F32(33);
  LinBNP pred_l1    { F32(34), F32(35), F32(36), F32(37) };
  LinBNP p_align    { F32(38), F32(39), F32(40), F32(41) };
  LinBNP p_attend   { F32(42), F32(43), F32(44), F32(45) };
  const float *pWhh = F32(46), *pWih = F32(47), *pbhh = F32(48), *pbih = F32(49);
  LinBNP s_align    { F32(50), F32(51), F32(52), F32(53) };
  LinBNP s_attend   { F32(54), F32(55), F32(56), F32(57) };
  const float *sWhh = F32(58), *sWih = F32(59), *sbhh = F32(60), *sbih = F32(61);

  // ---- workspace bump allocator (~2.0 GB) ----
  char* wsbase = (char*)d_ws;
  size_t off = 0;
  auto alloc = [&](size_t bytes) -> char* {
    off = (off + 255) & ~(size_t)255;
    char* p = wsbase + off;
    off += bytes;
    return p;
  };
  float* A     = (float*)alloc((size_t)NA * F * 4);
  float* ANEW  = (float*)alloc((size_t)NA * F * 4);
  float* BF    = (float*)alloc((size_t)NB * F * 4);
  float* NEIGH = (float*)alloc((size_t)NB * 3 * F * 4);
  float* BIG   = (float*)alloc((size_t)NB * 3 * F * 4);  // attend out / bond tmp
  float* CTX   = (float*)alloc((size_t)NA * 3 * F * 4);
  float* GI    = (float*)alloc((size_t)NA * 3 * F * 4);
  float* GH    = (float*)alloc((size_t)NA * 3 * F * 4);
  float* ATOMP = (float*)alloc((size_t)NA * 64 * 4);     // atom padded 39->64
  float* BONDP = (float*)alloc((size_t)NB * 32 * 4);     // bond padded 10->32
  float* WPA   = (float*)alloc((size_t)F * 64 * 4);      // atom_l1.W padded
  float* WPB   = (float*)alloc((size_t)F * 32 * 4);      // bond_l1.W padded
  float* SCORE = (float*)alloc((size_t)NB * 4);
  float* WGT   = (float*)alloc((size_t)NB * 4);
  unsigned* SMAX = (unsigned*)alloc((size_t)NA * 4);
  float* SSUM  = (float*)alloc((size_t)NA * 4);
  float* SA    = (float*)alloc((size_t)NM * F * 4);
  float* SANEW = (float*)alloc((size_t)NM * F * 4);
  float* CTXM  = (float*)alloc((size_t)NM * F * 4);
  float* GIM   = (float*)alloc((size_t)NM * 3 * F * 4);
  float* GHM   = (float*)alloc((size_t)NM * 3 * F * 4);
  float* PH    = (float*)alloc((size_t)NM * 512 * 4);
  float* STAT  = (float*)alloc((size_t)4 * 512 * 4);
  int*   BI0   = (int*)alloc((size_t)NB * 4);
  int*   BI1   = (int*)alloc((size_t)NB * 4);
  (void)ws_size; (void)n_in;

  split_idx_kernel<<<cdiv(NB, 256), 256, 0, stream>>>(bidx, BI0, BI1, NB);

  // ---- pad ragged-K operands so the GEMM path is branch-free b128 loads ----
  pad_kernel<<<cdiv((size_t)NA * 64, 256), 256, 0, stream>>>(atom, ATOMP, NA, 39, 64);
  pad_kernel<<<cdiv((size_t)F * 64, 256), 256, 0, stream>>>(atom_l1.W, WPA, F, 39, 64);
  pad_kernel<<<cdiv((size_t)NB * 32, 256), 256, 0, stream>>>(bond, BONDP, NB, 10, 32);
  pad_kernel<<<cdiv((size_t)F * 32, 256), 256, 0, stream>>>(bond_l1.W, WPB, F, 10, 32);

  // ---- preprocess ----
  run_linbn(stream, ATOMP, nullptr, NA, 64, F, WPA,         atom_l1, CTX, STAT, 1);
  run_linbn(stream, CTX,   nullptr, NA, F,  F, atom_l2.W,   atom_l2, A,   STAT, 1);
  run_linbn(stream, BONDP, nullptr, NB, 32, F, WPB,         bond_l1, BIG, STAT, 1);
  run_linbn(stream, BIG,   nullptr, NB, F,  F, bond_l2.W,   bond_l2, BF,  STAT, 1);

  // ---- first graph attention (neighbor = [b | na | mix], 384 wide) ----
  neigh_kernel<<<cdiv((size_t)NB * F, 256), 256, 0, stream>>>(A, BF, BI1, NEIGH,
                                                              (size_t)NB * F, F);
  run_scores(stream, A, BI0, F, NEIGH, nullptr, 3 * F, align, NB, BI0, NA,
             SCORE, WGT, STAT, SMAX, SSUM);
  run_linbn(stream, NEIGH, nullptr, NB, 3 * F, 3 * F, attend.W, attend, BIG, STAT, 0);
  run_ctx(stream, BIG, WGT, BI0, CTX, NB, NA, 3 * F, true);
  run_gru(stream, CTX, A, gWih, gbih, gWhh, gbhh, NA, 3 * F, F, GI, GH, ANEW);
  { float* t = A; A = ANEW; ANEW = t; }

  // ---- K-1 = 2 propagate rounds ----
  for (int it = 0; it < 2; ++it) {
    run_scores(stream, A, BI0, F, A, BI1, F, p_align, NB, BI0, NA,
               SCORE, WGT, STAT, SMAX, SSUM);
    run_linbn(stream, A, BI1, NB, F, F, p_attend.W, p_attend, BIG, STAT, 0);
    run_ctx(stream, BIG, WGT, BI0, CTX, NB, NA, F, true);
    run_gru(stream, CTX, A, pWih, pbih, pWhh, pbhh, NA, F, F, GI, GH, ANEW);
    float* t = A; A = ANEW; ANEW = t;
  }

  // ---- super node: sa = segment_sum(a, mol_index) ----
  run_ctx(stream, A, nullptr, molidx, SA, NA, NM, F, false);

  // ---- T = 3 super attention rounds ----
  for (int it = 0; it < 3; ++it) {
    run_scores(stream, SA, molidx, F, A, nullptr, F, s_align, NA, molidx, NM,
               SCORE, WGT, STAT, SMAX, SSUM);
    run_linbn(stream, A, nullptr, NA, F, F, s_attend.W, s_attend, BIG, STAT, 0);
    run_ctx(stream, BIG, WGT, molidx, CTXM, NA, NM, F, true);
    run_gru(stream, CTXM, SA, sWih, sbih, sWhh, sbhh, NM, F, F, GIM, GHM, SANEW);
    float* t = SA; SA = SANEW; SANEW = t;
  }

  // ---- prediction head ----
  run_linbn(stream, SA, nullptr, NM, F, 512, pred_l1.W, pred_l1, PH, STAT, 1);
  dot2_kernel<<<cdiv(NM, 8), 256, 0, stream>>>(PH, nullptr, 512, nullptr, nullptr, 0,
                                               predW, predB, (float*)d_out, nullptr, NM);
}
